// PolicyNet_18056042512532
// MI455X (gfx1250) — matrix operand order repaired and checked by reading, not compiled, not run on hardware
//
#include <hip/hip_runtime.h>
#include <math.h>

#define NN    48
#define MAXD  56
#define MD    9
#define GC1   128
#define GC2   64
#define AUXD  128
#define WPB   2
#define BTOT  8192

typedef _Float16 h16;
typedef __attribute__((ext_vector_type(16))) _Float16 v16h;
typedef __attribute__((ext_vector_type(8)))  _Float16 v8h;
typedef __attribute__((ext_vector_type(8)))  float    v8f;
typedef unsigned int u32;
typedef __attribute__((ext_vector_type(4))) u32 v4u;
typedef __attribute__((ext_vector_type(4))) int v4i;
typedef __attribute__((ext_vector_type(8))) int v8i;

union V16U { v16h v; v8h h[2]; };

struct Params {
  const int*   Nn;
  const float* adj;
  const float *W1,*b1,*W2,*b2,*asW,*asb,*atW,*atb;
  const float *l1W,*l1b,*l2W,*l2b;
  const float *f1,*f2,*s1,*s2,*e1,*e2;
  const float *t1W,*t1b,*t2W,*t2b;
  const float *v1W,*v1b,*v2W,*v2b,*v3W,*v3b;
  float* out;
};

static __device__ inline v8f wmma16(v16h a, v16h b, v8f c) {
  // D = A(16x32 f16) x B(32x16 f16) + C(16x16 f32)
  return __builtin_amdgcn_wmma_f32_16x16x32_f16(false, a, false, b, (short)0, c,
                                                false, false);
}

// CDNA5 hardware tanh (transcendental unit) instead of branchy libm tanhf.
static __device__ inline float fast_tanh(float x) {
  float y;
  asm volatile("v_tanh_f32 %0, %1" : "=v"(y) : "v"(x));
  return y;
}
static __device__ inline float lrelu(float x) { return x > 0.f ? x : 0.1f * x; }
static __device__ inline float sigm(float x)  { return 1.f / (1.f + __expf(-x)); }

// A fragment: 16x32 f16 tile from row-major LDS array (stride in halves).
static __device__ inline v16h ldsLoadA(const h16* base, int stride, int row0,
                                       int col0, int lane) {
  int m  = lane & 15;
  int hi = lane >> 4;              // lanes 16..31 hold K+8..15 / K+24..31
  const h16* p = base + (row0 + m) * stride + col0 + hi * 8;
  V16U u;
  u.h[0] = *(const v8h*)p;         // K = col0 + hi*8 + [0..7]
  u.h[1] = *(const v8h*)(p + 16);  // K = col0 + 16 + hi*8 + [0..7]
  return u.v;
}

// B fragment: 32x16 f16 tile from row-major [K][N] LDS array.
static __device__ inline v16h ldsLoadB(const h16* base, int stride, int krow0,
                                       int col0, int lane) {
  const h16* p = base + (krow0 + (lane >> 4) * 8) * stride + col0 + (lane & 15);  // N = col0 + lane%16, K = krow0 + hi*8 + [0..7], +16
  v16h r;
#pragma unroll
  for (int s = 0; s < 8; ++s) {
    r[s] = p[s * stride];
    r[s + 8] = p[(s + 16) * stride];
  }
  return r;
}

__global__ __launch_bounds__(WPB * 32)
void policy_kernel(Params P) {
  extern __shared__ char smem[];
  // block-shared f16 weight copies
  h16* sW2 = (h16*)smem;                 // 128 x 64
  h16* sWs = (h16*)(smem + 16384);       // 96 x 128  (rows 73..95 zero)
  h16* sWt = (h16*)(smem + 40960);       // 96 x 128
  const int tid  = threadIdx.x;
  const int wv   = tid >> 5;
  const int lane = tid & 31;
  const u32 woff = 65536u + (u32)wv * 49920u;   // per-wave LDS byte offset
  char* wbase = smem + woff;
  h16*   sA    = (h16*)wbase;              // [3*48][64]   adj_n f16, K-padded
  h16*   sB    = (h16*)(wbase + 18432);    // [64][128]    B operands
  h16*   sC    = (h16*)(wbase + 34816);    // [48][128]    A-side intermediates
  int*   sNode = (int*)(wbase + 47104);    // 64 ints
  float* sG    = (float*)(wbase + 47360);  // 128 f32
  float* sS    = (float*)(wbase + 47872);  // 512 f32 scratch

  const int b = blockIdx.x * WPB + wv;

  // ---- TDM: DMA the whole per-batch adj tile (3x48x48 f32 = 27648B) into the
  //      (currently dead) sB..sC LDS region in ONE tensor_load_to_lds.
  {
    unsigned long long ga =
        (unsigned long long)(uintptr_t)(P.adj + (size_t)b * 3 * NN * NN);
    v4u g0;
    g0.x = 1u;                                   // count=1 (valid descriptor)
    g0.y = woff + 18432u;                        // lds_addr (bytes)
    g0.z = (u32)ga;                              // global_addr[31:0]
    g0.w = (u32)(ga >> 32) | (2u << 30);         // global_addr[56:32] | type=2
    v8i g1;
    g1[0] = (2 << 16);                           // data_size = 4 bytes
    g1[1] = (NN << 16);                          // tensor_dim0 lo16 = 48
    g1[2] = (NN << 16);                          // dim0 hi=0 | tensor_dim1 lo16 = 48
    g1[3] = (NN << 16);                          // dim1 hi=0 | tile_dim0 = 48
    g1[4] = NN | (3 << 16);                      // tile_dim1 = 48, tile_dim2 = 3
    g1[5] = NN;                                  // tensor_dim0_stride = 48
    g1[6] = ((NN * NN) << 16);                   // stride0 hi=0 | stride1 lo16 = 2304
    g1[7] = 0;                                   // stride1 hi = 0
    v4i g2; g2[0] = 3; g2[1] = 0; g2[2] = 0; g2[3] = 0;   // tensor_dim2 = 3
    v4i g3; g3[0] = 0; g3[1] = 0; g3[2] = 0; g3[3] = 0;
#if __clang_major__ >= 23
    v8i gpad; gpad[0]=0;gpad[1]=0;gpad[2]=0;gpad[3]=0;gpad[4]=0;gpad[5]=0;gpad[6]=0;gpad[7]=0;
    __builtin_amdgcn_tensor_load_to_lds(g0, g1, g2, g3, gpad, 0);
#else
    __builtin_amdgcn_tensor_load_to_lds(g0, g1, g2, g3, 0);
#endif
  }

  // ---- stage shared weights (whole block) while TDM is in flight ----
  for (int i = tid; i < 128 * 64; i += blockDim.x) sW2[i] = (h16)P.W2[i];
  for (int i = tid; i < 96 * 128; i += blockDim.x) {
    int r = i >> 7, c = i & 127;
    float ws = (r < 73) ? P.asW[r * 128 + c] : 0.f;
    float wt = (r < 73) ? P.atW[r * 128 + c] : 0.f;
    sWs[i] = (h16)ws;
    sWt[i] = (h16)wt;
  }
  __syncthreads();

  // ---- node ids, zero gate accumulator ----
  for (int i = lane; i < MAXD; i += 32) sNode[i] = P.Nn[b * MAXD + i];
  for (int d = lane; d < 128; d += 32) sG[d] = 0.f;

  // ---- wait for TDM, then adj_n = adj * 1/(rowsum+1) -> f16 sA (cols>=48 zero)
  __builtin_amdgcn_s_wait_tensorcnt(0);
  const float* raw = (const float*)(wbase + 18432);
  for (int row = lane; row < 3 * NN; row += 32) {
    const float4* src = (const float4*)(raw + row * NN);
    float v[48];
    float s = 0.f;
#pragma unroll
    for (int q = 0; q < 12; ++q) {
      float4 x = src[q];
      v[4 * q + 0] = x.x; v[4 * q + 1] = x.y;
      v[4 * q + 2] = x.z; v[4 * q + 3] = x.w;
      s += x.x + x.y + x.z + x.w;
    }
    float sc = 1.f / (s + 1.f);
    h16* dst = sA + row * 64;
#pragma unroll
    for (int q = 0; q < 6; ++q) {
      v8h p;
#pragma unroll
      for (int e = 0; e < 8; ++e) p[e] = (h16)(v[8 * q + e] * sc);
      *(v8h*)(dst + 8 * q) = p;
    }
    v8h z = {};
    *(v8h*)(dst + 48) = z;
    *(v8h*)(dst + 56) = z;
  }

  // ---- h1 = onehot @ W1 + b1 (row gather of W1) -> sB[64][128], rows>=48 zero
  //      (overwrites the raw adj staging area -- adj already consumed above)
  for (int i = lane; i < 64 * 128; i += 32) {
    int r = i >> 7, c = i & 127;
    float v = 0.f;
    if (r < NN) v = P.W1[sNode[r] * GC1 + c] + P.b1[c];
    sB[i] = (h16)v;
  }

  // ---- GEMM1: h = tanh(sum_r adj_n[r] @ h1 + h1) -> sC[48][128]
  for (int mt = 0; mt < 3; ++mt)
    for (int nt = 0; nt < 8; ++nt) {
      int col = nt * 16 + (lane & 15);
      int rb  = mt * 16 + (lane >> 4) * 8;
      v8f acc;
#pragma unroll
      for (int v = 0; v < 8; ++v) acc[v] = (float)sB[(rb + v) * 128 + col];
#pragma unroll
      for (int r = 0; r < 3; ++r)
#pragma unroll
        for (int kt = 0; kt < 2; ++kt) {
          v16h a  = ldsLoadA(sA + r * NN * 64, 64, mt * 16, kt * 32, lane);
          v16h bb = ldsLoadB(sB, 128, kt * 32, nt * 16, lane);
          acc = wmma16(a, bb, acc);
        }
#pragma unroll
      for (int v = 0; v < 8; ++v)
        sC[(rb + v) * 128 + col] = (h16)fast_tanh(acc[v]);
    }

  // ---- GEMM2: h2 = h @ W2 + b2 -> sB rows 0..47, cols 0..63 (rows 48..63 stay 0)
  for (int mt = 0; mt < 3; ++mt)
    for (int nt = 0; nt < 4; ++nt) {
      int col = nt * 16 + (lane & 15);
      int rb  = mt * 16 + (lane >> 4) * 8;
      float bias = P.b2[col];
      v8f acc;
#pragma unroll
      for (int v = 0; v < 8; ++v) acc[v] = bias;
#pragma unroll
      for (int kt = 0; kt < 4; ++kt) {
        v16h a  = ldsLoadA(sC, 128, mt * 16, kt * 32, lane);
        v16h bb = ldsLoadB(sW2, 64, kt * 32, nt * 16, lane);
        acc = wmma16(a, bb, acc);
      }
#pragma unroll
      for (int v = 0; v < 8; ++v) sB[(rb + v) * 128 + col] = (h16)acc[v];
    }

  // ---- GEMM3: h_out = tanh(sum_r adj_n[r] @ h2 + h2) -> sC cols 0..63 (ann lo)
  for (int mt = 0; mt < 3; ++mt)
    for (int nt = 0; nt < 4; ++nt) {
      int col = nt * 16 + (lane & 15);
      int rb  = mt * 16 + (lane >> 4) * 8;
      v8f acc;
#pragma unroll
      for (int v = 0; v < 8; ++v) acc[v] = (float)sB[(rb + v) * 128 + col];
#pragma unroll
      for (int r = 0; r < 3; ++r)
#pragma unroll
        for (int kt = 0; kt < 2; ++kt) {
          v16h a  = ldsLoadA(sA + r * NN * 64, 64, mt * 16, kt * 32, lane);
          v16h bb = ldsLoadB(sB, 128, kt * 32, nt * 16, lane);
          acc = wmma16(a, bb, acc);
        }
#pragma unroll
      for (int v = 0; v < 8; ++v)
        sC[(rb + v) * 128 + col] = (h16)fast_tanh(acc[v]);
    }

  // ---- ann hi: onehot into sC cols 64..72, zeros to col 95
  for (int i = lane; i < NN * 32; i += 32) {
    int r = i >> 5, c = (i & 31) + 64;
    float v = (c < 73) ? ((sNode[r] == (c - 64)) ? 1.f : 0.f) : 0.f;
    sC[r * 128 + c] = (h16)v;
  }

  // ---- GEMM4: gate = sigmoid(ann@asW+asb) * tanh(ann@atW+atb); colsum -> sG
  for (int mt = 0; mt < 3; ++mt)
    for (int nt = 0; nt < 8; ++nt) {
      int col = nt * 16 + (lane & 15);
      float bs = P.asb[col], bt = P.atb[col];
      v8f as, at;
#pragma unroll
      for (int v = 0; v < 8; ++v) { as[v] = bs; at[v] = bt; }
#pragma unroll
      for (int kt = 0; kt < 3; ++kt) {
        v16h a  = ldsLoadA(sC, 128, mt * 16, kt * 32, lane);
        v16h b0 = ldsLoadB(sWs, 128, kt * 32, nt * 16, lane);
        as = wmma16(a, b0, as);
        v16h b1 = ldsLoadB(sWt, 128, kt * 32, nt * 16, lane);
        at = wmma16(a, b1, at);
      }
      float part = 0.f;
#pragma unroll
      for (int v = 0; v < 8; ++v) part += sigm(as[v]) * fast_tanh(at[v]);
      atomicAdd(&sG[col], part);   // two lanes share a column -> ds_add_f32
    }

  // ---- g = tanh(colsum) ----
  for (int d = lane; d < 128; d += 32) sG[d] = fast_tanh(sG[d]);

  // ---- hfeat = (g @ l1W + l1b) @ l2W + l2b ----
  for (int j = lane; j < 128; j += 32) {
    float s = P.l1b[j];
    for (int d = 0; d < 128; ++d) s += sG[d] * P.l1W[d * 128 + j];
    sS[j] = s;
  }
  float* hf = sS + 128;  // 64
  for (int k = lane; k < 64; k += 32) {
    float s = P.l2b[k];
    for (int j = 0; j < 128; ++j) s += sS[j] * P.l2W[j * 64 + k];
    hf[k] = s;
  }
  float* u = sS + 192;  // 32-wide hidden, reused per head

  // ---- head f: out1 = lrelu(hf@f1) @ f2 ----
  {
    float s = 0.f;
    for (int k = 0; k < 64; ++k) s += hf[k] * P.f1[k * 32 + lane];
    u[lane] = lrelu(s);
  }
  for (int j = lane; j < MAXD; j += 32) {
    float s = 0.f;
    for (int t = 0; t < 32; ++t) s += u[t] * P.f2[t * MAXD + j];
    sS[224 + j] = s;
  }
  // ---- head s: out2 ----
  {
    float s = 0.f;
    for (int k = 0; k < 64; ++k) s += hf[k] * P.s1[k * 32 + lane];
    u[lane] = lrelu(s);
  }
  for (int j = lane; j < MAXD; j += 32) {
    float s = 0.f;
    for (int t = 0; t < 32; ++t) s += u[t] * P.s2[t * MAXD + j];
    sS[288 + j] = s;
  }
  // ---- head e: oute (3) ----
  {
    float s = 0.f;
    for (int k = 0; k < 64; ++k) s += hf[k] * P.e1[k * 32 + lane];
    u[lane] = lrelu(s);
  }
  for (int j = lane; j < 3; j += 32) {
    float s = 0.f;
    for (int t = 0; t < 32; ++t) s += u[t] * P.e2[t * 3 + j];
    sS[344 + j] = s;
  }
  // ---- head t: outs (2) ----
  {
    float s = P.t1b[lane];
    for (int k = 0; k < 64; ++k) s += hf[k] * P.t1W[k * 32 + lane];
    u[lane] = lrelu(s);
  }
  for (int j = lane; j < 2; j += 32) {
    float s = P.t2b[j];
    for (int t = 0; t < 32; ++t) s += u[t] * P.t2W[t * 2 + j];
    sS[348 + j] = s;
  }
  // ---- head v: value ----
  {
    float s = P.v1b[lane];
    for (int k = 0; k < 64; ++k) s += hf[k] * P.v1W[k * 32 + lane];
    u[lane] = lrelu(s);
  }
  for (int j = lane; j < 16; j += 32) {
    float s = P.v2b[j];
    for (int t = 0; t < 32; ++t) s += u[t] * P.v2W[t * 16 + j];
    sS[360 + j] = lrelu(s);
  }
  if (lane == 0) {
    float s = P.v3b[0];
    for (int j = 0; j < 16; ++j) s += sS[360 + j] * P.v3W[j];
    sS[350] = s;
  }

  // ---- masked softmaxes ----
  if (lane == 0) {
    float m1 = -1e30f; int fi = 0;
    for (int j = 0; j < MAXD; ++j) {
      bool ex = (sNode[j] > 0) && (j < NN);
      float lg = (ex ? 0.f : -10000.f) + sS[224 + j];
      if (lg > m1) { m1 = lg; fi = j; }
    }
    float s1s = 0.f;
    for (int j = 0; j < MAXD; ++j) {
      bool ex = (sNode[j] > 0) && (j < NN);
      float lg = (ex ? 0.f : -10000.f) + sS[224 + j];
      s1s += __expf(lg - m1);
    }
    sS[352] = m1; sS[353] = 1.f / s1s; sNode[60] = fi;
    float m2 = -1e30f;
    for (int j = 0; j < MAXD; ++j) {
      bool ex = (sNode[j] > 0) && (j < NN);
      float msk = (ex || j >= NN) ? 1.f : 0.f;
      if (j == fi) msk = 0.f;
      float lg = (msk - 1.f) * 10000.f + sS[288 + j];
      m2 = fmaxf(m2, lg);
    }
    float s2s = 0.f;
    for (int j = 0; j < MAXD; ++j) {
      bool ex = (sNode[j] > 0) && (j < NN);
      float msk = (ex || j >= NN) ? 1.f : 0.f;
      if (j == fi) msk = 0.f;
      float lg = (msk - 1.f) * 10000.f + sS[288 + j];
      s2s += __expf(lg - m2);
    }
    sS[354] = m2; sS[355] = 1.f / s2s;
  }

  float m1 = sS[352], is1 = sS[353], m2 = sS[354], is2 = sS[355];
  int fi = sNode[60];
  float* outp = P.out + (size_t)b * 118;
  for (int j = lane; j < MAXD; j += 32) {
    bool ex = (sNode[j] > 0) && (j < NN);
    float lg1 = (ex ? 0.f : -10000.f) + sS[224 + j];
    outp[j] = __expf(lg1 - m1) * is1;
    float msk = (ex || j >= NN) ? 1.f : 0.f;
    if (j == fi) msk = 0.f;
    float lg2 = (msk - 1.f) * 10000.f + sS[288 + j];
    outp[56 + j] = __expf(lg2 - m2) * is2;
  }
  if (lane == 0) {
    float a0 = sS[344], a1 = sS[345], a2 = sS[346];
    float mm = fmaxf(a0, fmaxf(a1, a2));
    float e0 = __expf(a0 - mm), e1 = __expf(a1 - mm), e2 = __expf(a2 - mm);
    float inv = 1.f / (e0 + e1 + e2);
    outp[112] = e0 * inv; outp[113] = e1 * inv; outp[114] = e2 * inv;
    float t0 = sS[348], t1 = sS[349];
    float mt = fmaxf(t0, t1);
    float x0 = __expf(t0 - mt), x1 = __expf(t1 - mt);
    float iv = 1.f / (x0 + x1);
    outp[115] = x0 * iv; outp[116] = x1 * iv;
    outp[117] = sS[350];
  }
}

extern "C" void kernel_launch(void* const* d_in, const int* in_sizes, int n_in,
                              void* d_out, int out_size, void* d_ws, size_t ws_size,
                              hipStream_t stream) {
  (void)in_sizes; (void)n_in; (void)out_size; (void)d_ws; (void)ws_size;
  Params P;
  P.Nn  = (const int*)d_in[0];
  P.adj = (const float*)d_in[1];
  P.W1  = (const float*)d_in[2];   P.b1  = (const float*)d_in[3];
  P.W2  = (const float*)d_in[4];   P.b2  = (const float*)d_in[5];
  P.asW = (const float*)d_in[6];   P.asb = (const float*)d_in[7];
  P.atW = (const float*)d_in[8];   P.atb = (const float*)d_in[9];
  P.l1W = (const float*)d_in[10];  P.l1b = (const float*)d_in[11];
  P.l2W = (const float*)d_in[12];  P.l2b = (const float*)d_in[13];
  P.f1  = (const float*)d_in[14];  P.f2  = (const float*)d_in[15];
  P.s1  = (const float*)d_in[16];  P.s2  = (const float*)d_in[17];
  P.e1  = (const float*)d_in[18];  P.e2  = (const float*)d_in[19];
  P.t1W = (const float*)d_in[20];  P.t1b = (const float*)d_in[21];
  P.t2W = (const float*)d_in[22];  P.t2b = (const float*)d_in[23];
  P.v1W = (const float*)d_in[24];  P.v1b = (const float*)d_in[25];
  P.v2W = (const float*)d_in[26];  P.v2b = (const float*)d_in[27];
  P.v3W = (const float*)d_in[28];  P.v3b = (const float*)d_in[29];
  P.out = (float*)d_out;

  const size_t smem = 65536 + (size_t)WPB * 49920;  // 165,376 B dynamic LDS
  dim3 grid(BTOT / WPB), block(WPB * 32);
  hipLaunchKernelGGL(policy_kernel, grid, block, smem, stream, P);
}
